// PoseGNN_62483184222219
// MI455X (gfx1250) — matrix operand order repaired and checked by reading, not compiled, not run on hardware
//
#include <hip/hip_runtime.h>
#include <stdint.h>

// ---------------------------------------------------------------------------
// PoseGNN on MI455X (gfx1250):
//   3x GraphConv = fused dual GEMM relu(agg@Wrel + h@Wroot + b) using
//   v_wmma_f32_16x16x32_bf16; f32 operands are converted to bf16 in-register
//   with a 1-instruction v_perm_b32 pack (round-half-up) so conversion VALU
//   co-executes under the WMMAs. f32-atomic edge scatter-add (L2-resident),
//   small VALU head.
// ---------------------------------------------------------------------------

#define N_NODES 16384
#define N_EDGES 65536

typedef __attribute__((ext_vector_type(16))) __bf16       v16bf;
typedef __attribute__((ext_vector_type(8)))  float        v8f;
typedef __attribute__((ext_vector_type(8)))  unsigned int v8u;

union Frag {
    v8u   u;
    v16bf b;
};

// ---- f32 pair -> packed bf16x2: 2x v_add + 1x v_perm_b32 ------------------
__device__ __forceinline__ uint32_t pack2bf(float lo, float hi) {
    uint32_t ul = __float_as_uint(lo) + 0x8000u;   // round-half-up to bf16
    uint32_t uh = __float_as_uint(hi) + 0x8000u;
    // result bytes {uh[3],uh[2],ul[3],ul[2]}  ->  (bf16(hi)<<16)|bf16(lo)
    return __builtin_amdgcn_perm(uh, ul, 0x07060302u);
}

// ---- scalar f32 -> bf16 (RNE) for cold paths ------------------------------
__device__ __forceinline__ uint16_t f32_to_bf16(float f) {
    uint32_t u = __float_as_uint(f);
    u += 0x7FFFu + ((u >> 16) & 1u);
    return (uint16_t)(u >> 16);
}

// W[K,N] f32 (row-major) -> Wt[N,K] bf16 (row-major): K contiguous per row.
__global__ void transpose_bf16_kernel(const float* __restrict__ W,
                                      uint16_t* __restrict__ Wt,
                                      int K, int N) {
    int i = blockIdx.x * blockDim.x + threadIdx.x;
    if (i >= K * N) return;
    int k = i / N;
    int n = i - k * N;
    Wt[(size_t)n * K + k] = f32_to_bf16(W[i]);
}

// ---- edge aggregation: agg[dst] += x[src] (f32 atomics, float4 rows) ------
__global__ void scatter_add_kernel(const float* __restrict__ x,
                                   const int* __restrict__ src,
                                   const int* __restrict__ dst,
                                   float* __restrict__ agg, int F) {
    int e = blockIdx.x;
    int s = src[e];
    int d = dst[e];
    const float4* __restrict__ xr = (const float4*)(x + (size_t)s * F);
    float* __restrict__ ar = agg + (size_t)d * F;
    int f4 = F >> 2;
    for (int i = threadIdx.x; i < f4; i += blockDim.x) {
        float4 v = xr[i];
        atomicAdd(ar + 4 * i + 0, v.x);
        atomicAdd(ar + 4 * i + 1, v.y);
        atomicAdd(ar + 4 * i + 2, v.z);
        atomicAdd(ar + 4 * i + 3, v.w);
    }
}

// ---------------------------------------------------------------------------
// One K-pass of the dual GEMM: acc += A @ Wt^T for a 32x64 wave tile.
//   A : [M,K] row-major, f32 (converted in-register) or bf16
//   Wt: [N,K] bf16 (K contiguous)
// A-fragment layout (16-bit 16x32, ISA 7.12.2): lane row = lane%16; lanes
// 0-15 carry K runs {0..7,16..23}, lanes 16-31 carry {8..15,24..31}.
// B-fragment 32x16: lane col = lane%16; lanes 0-15 K={0..7,16..23}, lanes 16-31
// K={8..15,24..31}, packed 2/VGPR along K.
// ---------------------------------------------------------------------------
template <int K, int NT, bool F32>
__device__ __forceinline__ void gemm_pass(const void* __restrict__ A,
                                          const uint16_t* __restrict__ Wt,
                                          v8f acc[2][NT],
                                          int mBase, int nBase,
                                          int half, int nl) {
    const float*    af = (const float*)A;
    const uint16_t* ah = (const uint16_t*)A;
    for (int kk = 0; kk < K; kk += 32) {
        Frag a[2];
#pragma unroll
        for (int mi = 0; mi < 2; ++mi) {
            size_t row = (size_t)(mBase + 16 * mi + nl);
            const int k0 = kk + half * 8;
            const int k1 = kk + 16 + half * 8;
            if constexpr (F32) {
                const float* __restrict__ r = af + row * K;
                if (kk + 32 < K)
                    __builtin_prefetch((const void*)(r + kk + 32), 0, 1);
                float4 f0 = *(const float4*)(r + k0);
                float4 f1 = *(const float4*)(r + k0 + 4);
                float4 f2 = *(const float4*)(r + k1);
                float4 f3 = *(const float4*)(r + k1 + 4);
                a[mi].u[0] = pack2bf(f0.x, f0.y);
                a[mi].u[1] = pack2bf(f0.z, f0.w);
                a[mi].u[2] = pack2bf(f1.x, f1.y);
                a[mi].u[3] = pack2bf(f1.z, f1.w);
                a[mi].u[4] = pack2bf(f2.x, f2.y);
                a[mi].u[5] = pack2bf(f2.z, f2.w);
                a[mi].u[6] = pack2bf(f3.x, f3.y);
                a[mi].u[7] = pack2bf(f3.z, f3.w);
            } else {
                const uint16_t* __restrict__ r = ah + row * K;
                if (kk + 32 < K)
                    __builtin_prefetch((const void*)(r + kk + 32), 0, 1);
#pragma unroll
                for (int j = 0; j < 4; ++j)
                    a[mi].u[j] = *(const unsigned int*)(r + k0 + 2 * j);
#pragma unroll
                for (int j = 0; j < 4; ++j)
                    a[mi].u[4 + j] = *(const unsigned int*)(r + k1 + 2 * j);
            }
        }
#pragma unroll
        for (int t = 0; t < NT; ++t) {
            const uint16_t* __restrict__ brow =
                Wt + (size_t)(nBase + 16 * t + nl) * K;
            Frag b;
#pragma unroll
            for (int j = 0; j < 8; ++j)
                b.u[j] = *(const unsigned int*)(brow + kk + (j >> 2) * 16 + half * 8 + 2 * (j & 3));
            acc[0][t] = __builtin_amdgcn_wmma_f32_16x16x32_bf16(
                false, a[0].b, false, b.b, (short)0, acc[0][t], false, false);
            acc[1][t] = __builtin_amdgcn_wmma_f32_16x16x32_bf16(
                false, a[1].b, false, b.b, (short)0, acc[1][t], false, false);
        }
    }
}

// Fused dual GEMM:  out = relu(A0 @ W0 + A1 @ W1 + bias), plus optional bf16
// copy of the output for the next layer's GEMM input.
// Wave tile: 32x64 (2 m-tiles x 4 n-tiles). 8 waves/block -> 256x64 block
// tile. grid = (M/256, N/64).
template <int K, int N, bool A0F32, bool A1F32>
__global__ __launch_bounds__(256)
void gemm2_bias_relu_kernel(const void* __restrict__ A0,
                            const uint16_t* __restrict__ W0t,
                            const void* __restrict__ A1,
                            const uint16_t* __restrict__ W1t,
                            const float* __restrict__ bias,
                            float* __restrict__ out,
                            uint16_t* __restrict__ out_b16) {
    constexpr int NT = 4;                       // 4 n-tiles = 64 cols / wave
    const int wid   = threadIdx.x >> 5;
    const int lane  = threadIdx.x & 31;
    const int half  = lane >> 4;
    const int nl    = lane & 15;
    const int mBase = (blockIdx.x * 16 + wid * 2) * 16;   // 32 rows / wave
    const int nBase = blockIdx.y * (NT * 16);

    v8f acc[2][NT];
    const v8f zero = {0.f, 0.f, 0.f, 0.f, 0.f, 0.f, 0.f, 0.f};
#pragma unroll
    for (int mi = 0; mi < 2; ++mi)
#pragma unroll
        for (int t = 0; t < NT; ++t) acc[mi][t] = zero;

    gemm_pass<K, NT, A0F32>(A0, W0t, acc, mBase, nBase, half, nl);
    gemm_pass<K, NT, A1F32>(A1, W1t, acc, mBase, nBase, half, nl);

    // C/D layout: VGPR r -> M = base + half*8 + r, N = nBase + 16*t + nl
#pragma unroll
    for (int mi = 0; mi < 2; ++mi) {
#pragma unroll
        for (int t = 0; t < NT; ++t) {
            int n = nBase + 16 * t + nl;
            float bv = bias[n];
#pragma unroll
            for (int r = 0; r < 8; ++r) {
                int m = mBase + 16 * mi + half * 8 + r;
                float v = acc[mi][t][r] + bv;
                v = v > 0.f ? v : 0.f;
                out[(size_t)m * N + n] = v;
                if (out_b16) out_b16[(size_t)m * N + n] = f32_to_bf16(v);
            }
        }
    }
}

// ---- heads: pos = h@Wp+bp, ori = normalize(h@Wo+bo, eps=1e-12) ------------
__global__ void heads_kernel(const float* __restrict__ h,
                             const float* __restrict__ Wp,
                             const float* __restrict__ bp,
                             const float* __restrict__ Wo,
                             const float* __restrict__ bo,
                             float* __restrict__ outPos,
                             float* __restrict__ outOri) {
    int node = blockIdx.x * blockDim.x + threadIdx.x;
    if (node >= N_NODES) return;
    const float* __restrict__ hr = h + (size_t)node * 64;
    float p0 = bp[0], p1 = bp[1], p2 = bp[2];
    float o0 = bo[0], o1 = bo[1], o2 = bo[2], o3 = bo[3];
#pragma unroll 8
    for (int k = 0; k < 64; ++k) {
        float hv = hr[k];
        p0 += hv * Wp[k * 3 + 0];
        p1 += hv * Wp[k * 3 + 1];
        p2 += hv * Wp[k * 3 + 2];
        o0 += hv * Wo[k * 4 + 0];
        o1 += hv * Wo[k * 4 + 1];
        o2 += hv * Wo[k * 4 + 2];
        o3 += hv * Wo[k * 4 + 3];
    }
    float nrm = sqrtf(o0 * o0 + o1 * o1 + o2 * o2 + o3 * o3);
    float inv = 1.0f / fmaxf(nrm, 1e-12f);
    outPos[node * 3 + 0] = p0;
    outPos[node * 3 + 1] = p1;
    outPos[node * 3 + 2] = p2;
    outOri[node * 4 + 0] = o0 * inv;
    outOri[node * 4 + 1] = o1 * inv;
    outOri[node * 4 + 2] = o2 * inv;
    outOri[node * 4 + 3] = o3 * inv;
}

// ---------------------------------------------------------------------------
extern "C" void kernel_launch(void* const* d_in, const int* in_sizes, int n_in,
                              void* d_out, int out_size, void* d_ws,
                              size_t ws_size, hipStream_t stream) {
    const float* x       = (const float*)d_in[0];
    const int*   ei      = (const int*)d_in[1];
    const float* W1_rel  = (const float*)d_in[2];
    const float* b1      = (const float*)d_in[3];
    const float* W1_root = (const float*)d_in[4];
    const float* W2_rel  = (const float*)d_in[5];
    const float* b2      = (const float*)d_in[6];
    const float* W2_root = (const float*)d_in[7];
    const float* W3_rel  = (const float*)d_in[8];
    const float* b3      = (const float*)d_in[9];
    const float* W3_root = (const float*)d_in[10];
    const float* Wp      = (const float*)d_in[11];
    const float* bp      = (const float*)d_in[12];
    const float* Wo      = (const float*)d_in[13];
    const float* bo      = (const float*)d_in[14];

    const int* src = ei;
    const int* dst = ei + N_EDGES;

    // ---- workspace carve-up ----
    char* ws = (char*)d_ws;
    size_t off = 0;
    float* agg_f = (float*)(ws + off);        off += (size_t)N_NODES * 2048 * 4; // 128MB, reused per layer
    float* h1_f  = (float*)(ws + off);        off += (size_t)N_NODES * 256 * 4;  // 16MB
    uint16_t* h1_b = (uint16_t*)(ws + off);   off += (size_t)N_NODES * 256 * 2;  // 8MB
    float* h2_f  = (float*)(ws + off);        off += (size_t)N_NODES * 128 * 4;  // 8MB
    uint16_t* h2_b = (uint16_t*)(ws + off);   off += (size_t)N_NODES * 128 * 2;  // 4MB
    float* h3_f  = (float*)(ws + off);        off += (size_t)N_NODES * 64 * 4;   // 4MB
    uint16_t* w1r = (uint16_t*)(ws + off);    off += (size_t)2048 * 256 * 2;
    uint16_t* w1t = (uint16_t*)(ws + off);    off += (size_t)2048 * 256 * 2;
    uint16_t* w2r = (uint16_t*)(ws + off);    off += (size_t)256 * 128 * 2;
    uint16_t* w2t = (uint16_t*)(ws + off);    off += (size_t)256 * 128 * 2;
    uint16_t* w3r = (uint16_t*)(ws + off);    off += (size_t)128 * 64 * 2;
    uint16_t* w3t = (uint16_t*)(ws + off);    off += (size_t)128 * 64 * 2;

    auto cdiv = [](int a, int b) { return (a + b - 1) / b; };

    // ---- weight prep (transpose + bf16) ----
    transpose_bf16_kernel<<<cdiv(2048 * 256, 256), 256, 0, stream>>>(W1_rel, w1r, 2048, 256);
    transpose_bf16_kernel<<<cdiv(2048 * 256, 256), 256, 0, stream>>>(W1_root, w1t, 2048, 256);
    transpose_bf16_kernel<<<cdiv(256 * 128, 256), 256, 0, stream>>>(W2_rel, w2r, 256, 128);
    transpose_bf16_kernel<<<cdiv(256 * 128, 256), 256, 0, stream>>>(W2_root, w2t, 256, 128);
    transpose_bf16_kernel<<<cdiv(128 * 64, 256), 256, 0, stream>>>(W3_rel, w3r, 128, 64);
    transpose_bf16_kernel<<<cdiv(128 * 64, 256), 256, 0, stream>>>(W3_root, w3t, 128, 64);

    // ---- layer 1 (K=2048 -> N=256): A0 = agg (f32), A1 = x (f32) ----
    hipMemsetAsync(agg_f, 0, (size_t)N_NODES * 2048 * 4, stream);
    scatter_add_kernel<<<N_EDGES, 256, 0, stream>>>(x, src, dst, agg_f, 2048);
    gemm2_bias_relu_kernel<2048, 256, true, true>
        <<<dim3(N_NODES / 256, 256 / 64), 256, 0, stream>>>(
            agg_f, w1r, x, w1t, b1, h1_f, h1_b);

    // ---- layer 2 (K=256 -> N=128): A0 = agg (f32), A1 = h1 (bf16) ----
    hipMemsetAsync(agg_f, 0, (size_t)N_NODES * 256 * 4, stream);
    scatter_add_kernel<<<N_EDGES, 64, 0, stream>>>(h1_f, src, dst, agg_f, 256);
    gemm2_bias_relu_kernel<256, 128, true, false>
        <<<dim3(N_NODES / 256, 128 / 64), 256, 0, stream>>>(
            agg_f, w2r, h1_b, w2t, b2, h2_f, h2_b);

    // ---- layer 3 (K=128 -> N=64): A0 = agg (f32), A1 = h2 (bf16) ----
    hipMemsetAsync(agg_f, 0, (size_t)N_NODES * 128 * 4, stream);
    scatter_add_kernel<<<N_EDGES, 32, 0, stream>>>(h2_f, src, dst, agg_f, 128);
    gemm2_bias_relu_kernel<128, 64, true, false>
        <<<dim3(N_NODES / 256, 64 / 64), 256, 0, stream>>>(
            agg_f, w3r, h2_b, w3t, b3, h3_f, (uint16_t*)nullptr);

    // ---- heads ----
    float* outPos = (float*)d_out;
    float* outOri = (float*)d_out + (size_t)N_NODES * 3;
    heads_kernel<<<cdiv(N_NODES, 256), 256, 0, stream>>>(h3_f, Wp, bp, Wo, bo,
                                                         outPos, outOri);
}
